// MultiHeadCrossAttention_58222576665152
// MI455X (gfx1250) — matrix operand order repaired and checked by reading, not compiled, not run on hardware
//
#include <hip/hip_runtime.h>

// ---------------------------------------------------------------------------
// MI455X / gfx1250 multi-head cross-attention, bf16 WMMA pipeline.
//   B=2, Tq=Tk=2048, D=1024, H=16, Dh=64
// ---------------------------------------------------------------------------

typedef __attribute__((ext_vector_type(16))) __bf16    v16bf;
typedef __attribute__((ext_vector_type(8)))  float     v8f;
typedef __attribute__((ext_vector_type(4)))  float     f32x4;
typedef __attribute__((ext_vector_type(4)))  unsigned  u32x4;

union BF16Frag {
  v16bf  v;
  __bf16 b[16];
  u32x4  q4[2];
};

__device__ __forceinline__ v8f zero8() {
  v8f z = {0.f, 0.f, 0.f, 0.f, 0.f, 0.f, 0.f, 0.f};
  return z;
}

__device__ __forceinline__ v8f wmma_bf16(const BF16Frag& a, const BF16Frag& b, v8f c) {
  return __builtin_amdgcn_wmma_f32_16x16x32_bf16(
      false, a.v, false, b.v, (short)0, c, false, false);
}

// ---------------------------------------------------------------------------
// GEMM: C_bf16[M,N] = A_f32[M,K] @ B_f32[K,N]
// 256 threads = 8 waves; wave w owns rows [m0,m0+16), all waves share n0..n0+63.
// B tile (32x64) staged once per block into LDS as bf16, TRANSPOSED [n][k] so
// fragment reads are two ds_load_b128 per tile. Double-buffered, 1 barrier/step.
// Requires M % 128 == 0, N % 64 == 0, K % 32 == 0 (true for all launches here).
// ---------------------------------------------------------------------------
__global__ __launch_bounds__(256)
void gemm_f32f32_bf16(const float* __restrict__ A, const float* __restrict__ Bm,
                      __bf16* __restrict__ C, int M, int N, int K) {
  const int tid  = threadIdx.x;
  const int lane = tid & 31, wave = tid >> 5;
  const int hi   = lane >> 4, l15 = lane & 15;
  const int m0   = (blockIdx.y * 8 + wave) * 16;
  const int n0   = blockIdx.x * 64;

  __shared__ __bf16 ldsB[2][64][40];           // [buf][n][k], 80B row stride

  const int skk = tid >> 3;                    // staging: k row 0..31
  const int sc8 = (tid & 7) * 8;               // staging: col base 0..56

  v8f acc[4];
#pragma unroll
  for (int t = 0; t < 4; ++t) acc[t] = zero8();

  const float* arow = A + (unsigned)(m0 + l15) * (unsigned)K + (hi << 3);

  int p = 0;
  for (int k0 = 0; k0 < K; k0 += 32, p ^= 1) {
    { // cooperative stage of B tile, f32 -> bf16 (native cvt), transposed
      const float* brow = Bm + (unsigned)(k0 + skk) * (unsigned)N + (n0 + sc8);
      f32x4 v0 = *(const f32x4*)brow;
      f32x4 v1 = *(const f32x4*)(brow + 4);
      float vals[8] = {v0[0], v0[1], v0[2], v0[3], v1[0], v1[1], v1[2], v1[3]};
#pragma unroll
      for (int j = 0; j < 8; ++j) ldsB[p][sc8 + j][skk] = (__bf16)vals[j];
    }
    __syncthreads();

    BF16Frag a;   // A fragment: 4x f32x4 vector loads + native bf16 converts
    {
      const float* ap = arow + k0;
      f32x4 fa = *(const f32x4*)(ap);
      f32x4 fb = *(const f32x4*)(ap + 4);
      f32x4 fc = *(const f32x4*)(ap + 16);
      f32x4 fd = *(const f32x4*)(ap + 20);
      a.b[0] = (__bf16)fa[0]; a.b[1] = (__bf16)fa[1];
      a.b[2] = (__bf16)fa[2]; a.b[3] = (__bf16)fa[3];
      a.b[4] = (__bf16)fb[0]; a.b[5] = (__bf16)fb[1];
      a.b[6] = (__bf16)fb[2]; a.b[7] = (__bf16)fb[3];
      a.b[8]  = (__bf16)fc[0]; a.b[9]  = (__bf16)fc[1];
      a.b[10] = (__bf16)fc[2]; a.b[11] = (__bf16)fc[3];
      a.b[12] = (__bf16)fd[0]; a.b[13] = (__bf16)fd[1];
      a.b[14] = (__bf16)fd[2]; a.b[15] = (__bf16)fd[3];
    }
#pragma unroll
    for (int t = 0; t < 4; ++t) {
      BF16Frag bf;  // B fragment: halves 0..7 <-> k = hi*8 + 0..7, 8..15 <-> k = 16 + hi*8 + 0..7
      const __bf16* bp = &ldsB[p][t * 16 + l15][hi << 3];
      bf.q4[0] = *(const u32x4*)bp;
      bf.q4[1] = *(const u32x4*)(bp + 16);
      acc[t] = wmma_bf16(a, bf, acc[t]);
    }
  }
#pragma unroll
  for (int t = 0; t < 4; ++t) {
    const int n = n0 + t * 16 + l15;
#pragma unroll
    for (int r = 0; r < 8; ++r) {
      const int m = m0 + r + (hi << 3);
      C[(unsigned)m * (unsigned)N + n] = (__bf16)acc[t][r];
    }
  }
}

// ---------------------------------------------------------------------------
// GEMM: C_f32[M,N] = A_bf16[M,K] @ B_f32[K,N]   (output projection)
// ---------------------------------------------------------------------------
__global__ __launch_bounds__(256)
void gemm_bf16f32_f32(const __bf16* __restrict__ A, const float* __restrict__ Bm,
                      float* __restrict__ C, int M, int N, int K) {
  const int tid  = threadIdx.x;
  const int lane = tid & 31, wave = tid >> 5;
  const int hi   = lane >> 4, l15 = lane & 15;
  const int m0   = (blockIdx.y * 8 + wave) * 16;
  const int n0   = blockIdx.x * 64;

  __shared__ __bf16 ldsB[2][64][40];

  const int skk = tid >> 3;
  const int sc8 = (tid & 7) * 8;

  v8f acc[4];
#pragma unroll
  for (int t = 0; t < 4; ++t) acc[t] = zero8();

  const __bf16* arow = A + (unsigned)(m0 + l15) * (unsigned)K + (hi << 3);

  int p = 0;
  for (int k0 = 0; k0 < K; k0 += 32, p ^= 1) {
    {
      const float* brow = Bm + (unsigned)(k0 + skk) * (unsigned)N + (n0 + sc8);
      f32x4 v0 = *(const f32x4*)brow;
      f32x4 v1 = *(const f32x4*)(brow + 4);
      float vals[8] = {v0[0], v0[1], v0[2], v0[3], v1[0], v1[1], v1[2], v1[3]};
#pragma unroll
      for (int j = 0; j < 8; ++j) ldsB[p][sc8 + j][skk] = (__bf16)vals[j];
    }
    __syncthreads();

    BF16Frag a;   // bf16 A fragment: two 128-bit loads
    a.q4[0] = *(const u32x4*)(arow + k0);
    a.q4[1] = *(const u32x4*)(arow + k0 + 16);

#pragma unroll
    for (int t = 0; t < 4; ++t) {
      BF16Frag bf;
      const __bf16* bp = &ldsB[p][t * 16 + l15][hi << 3];
      bf.q4[0] = *(const u32x4*)bp;
      bf.q4[1] = *(const u32x4*)(bp + 16);
      acc[t] = wmma_bf16(a, bf, acc[t]);
    }
  }
#pragma unroll
  for (int t = 0; t < 4; ++t) {
    const int n = n0 + t * 16 + l15;
#pragma unroll
    for (int r = 0; r < 8; ++r) {
      const int m = m0 + r + (hi << 3);
      C[(unsigned)m * (unsigned)N + n] = acc[t][r];
    }
  }
}

// ---------------------------------------------------------------------------
// Flash attention with fused scale + bias + mask.
// One wave per (h, q-tile, b); h fastest so per-batch mask stays L2-resident
// across heads. V tiles are reused by all 128 q-tiles of a (b,h) -> L2-cached.
// ---------------------------------------------------------------------------
__global__ __launch_bounds__(32)
void attention_flash(const __bf16* __restrict__ Qp,
                     const __bf16* __restrict__ KVp,
                     const float* __restrict__ bias,
                     const int* __restrict__ mask,
                     __bf16* __restrict__ Op,
                     int Tq, int Tk) {
  const int lane = threadIdx.x & 31;
  const int hi   = lane >> 4;
  const int l15  = lane & 15;
  const int h    = blockIdx.x;
  const int qt   = blockIdx.y;
  const int b    = blockIdx.z;
  const int q0   = qt * 16;
  const float NEG_INF = -__builtin_inff();

  __shared__ __bf16 pLds[16][40];            // 16x32 P tile, 80B row stride

  // Q fragments: 16 rows x Dh=64 -> two 16x32 A-fragments (128-bit loads)
  BF16Frag qa[2];
  {
    const __bf16* qrow = Qp + (unsigned)(b * Tq + q0 + l15) * 1024u + h * 64;
    qa[0].q4[0] = *(const u32x4*)(qrow + (hi << 3));
    qa[0].q4[1] = *(const u32x4*)(qrow + 16 + (hi << 3));
    qa[1].q4[0] = *(const u32x4*)(qrow + 32 + (hi << 3));
    qa[1].q4[1] = *(const u32x4*)(qrow + 48 + (hi << 3));
  }

  float mrow[8], lrow[8];
  v8f o[4];
#pragma unroll
  for (int r = 0; r < 8; ++r) { mrow[r] = NEG_INF; lrow[r] = 0.f; }
#pragma unroll
  for (int t = 0; t < 4; ++t) o[t] = zero8();

  const float scale = 0.125f;                // 1/sqrt(Dh=64)
  const float* biasBase = bias + (unsigned)(h * Tq + q0) * (unsigned)Tk;
  const int*   maskBase = mask + (unsigned)(b * Tq + q0) * (unsigned)Tk;

  for (int kt = 0; kt < Tk; kt += 32) {
    if (kt + 32 < Tk) {                      // global_prefetch_b8 next bias/mask
      __builtin_prefetch(biasBase + (unsigned)l15 * (unsigned)Tk + kt + 32, 0, 1);
      __builtin_prefetch(maskBase + (unsigned)l15 * (unsigned)Tk + kt + 32, 0, 1);
    }

    // ---- S = Q @ K^T : two 16-key tiles, Dh split into 2 k-steps ----
    v8f s0 = zero8(), s1 = zero8();
    const __bf16* krow0 = KVp + (unsigned)(b * Tk + kt + l15)      * 2048u + h * 64;
    const __bf16* krow1 = KVp + (unsigned)(b * Tk + kt + 16 + l15) * 2048u + h * 64;
#pragma unroll
    for (int s = 0; s < 2; ++s) {
      BF16Frag b0, b1;  // halves 0..7 <-> dh = s*32 + hi*8 + 0..7, 8..15 <-> dh = s*32 + 16 + hi*8 + 0..7
      b0.q4[0] = *(const u32x4*)(krow0 + s * 32 + (hi << 3));
      b0.q4[1] = *(const u32x4*)(krow0 + s * 32 + (hi << 3) + 16);
      b1.q4[0] = *(const u32x4*)(krow1 + s * 32 + (hi << 3));
      b1.q4[1] = *(const u32x4*)(krow1 + s * 32 + (hi << 3) + 16);
      s0 = wmma_bf16(qa[s], b0, s0);
      s1 = wmma_bf16(qa[s], b1, s1);
    }

    // ---- fused scale + bias + mask + online softmax ----
#pragma unroll
    for (int r = 0; r < 8; ++r) {
      const int qm = r + (hi << 3);
      const unsigned boff = (unsigned)qm * (unsigned)Tk + kt;
      float v0 = s0[r] * scale + biasBase[boff + l15];
      float v1 = s1[r] * scale + biasBase[boff + 16 + l15];
      if (maskBase[boff + l15]      == 0) v0 = NEG_INF;
      if (maskBase[boff + 16 + l15] == 0) v1 = NEG_INF;

      float mx = fmaxf(v0, v1);
#pragma unroll
      for (int off = 1; off < 16; off <<= 1) mx = fmaxf(mx, __shfl_xor(mx, off, 16));
      const float mNew = fmaxf(mrow[r], mx);

      float sc, p0, p1;
      if (mNew == NEG_INF) { sc = 1.f; p0 = 0.f; p1 = 0.f; }
      else {
        sc = __expf(mrow[r] - mNew);
        p0 = __expf(v0 - mNew);
        p1 = __expf(v1 - mNew);
      }
      float ps = p0 + p1;
#pragma unroll
      for (int off = 1; off < 16; off <<= 1) ps += __shfl_xor(ps, off, 16);

      lrow[r] = lrow[r] * sc + ps;
      mrow[r] = mNew;
#pragma unroll
      for (int t = 0; t < 4; ++t) o[t][r] *= sc;

      pLds[qm][l15]      = (__bf16)p0;     // stage P for cross-lane transpose
      pLds[qm][16 + l15] = (__bf16)p1;
    }
    __syncthreads();                        // single-wave WG -> cheap

    // ---- O += P(16x32) @ V(32x64) ----
    BF16Frag pa;                            // A-fragment reload: two b128 ds loads
    {
      const __bf16* pr = &pLds[l15][hi << 3];
      pa.q4[0] = *(const u32x4*)pr;         // k = hi*8 + 0..7
      pa.q4[1] = *(const u32x4*)(pr + 16);  // k = 16 + hi*8 + 0..7
    }
    const __bf16* vbase = KVp + (unsigned)(b * Tk + kt) * 2048u + 1024 + h * 64;
#pragma unroll
    for (int t = 0; t < 4; ++t) {
      BF16Frag vb;                          // true column gather (L2-resident V)
#pragma unroll
      for (int i = 0; i < 16; ++i) {
        const int key = (i & 7) + (hi << 3) + ((i >> 3) << 4);
        vb.b[i] = vbase[(unsigned)key * 2048u + t * 16 + l15];
      }
      o[t] = wmma_bf16(pa, vb, o[t]);
    }
    __syncthreads();
  }

  // ---- normalize and write attention output (bf16) ----
#pragma unroll
  for (int r = 0; r < 8; ++r) {
    const float inv = lrow[r] > 0.f ? 1.f / lrow[r] : 0.f;
    const int qm = r + (hi << 3);
    __bf16* orow = Op + (unsigned)(b * Tq + q0 + qm) * 1024u + h * 64;
#pragma unroll
    for (int t = 0; t < 4; ++t) orow[t * 16 + l15] = (__bf16)(o[t][r] * inv);
  }
}

// ---------------------------------------------------------------------------
// Launch
// ---------------------------------------------------------------------------
extern "C" void kernel_launch(void* const* d_in, const int* in_sizes, int n_in,
                              void* d_out, int out_size, void* d_ws, size_t ws_size,
                              hipStream_t stream) {
  (void)in_sizes; (void)n_in; (void)out_size; (void)ws_size;

  constexpr int Bb = 2, Tq = 2048, Tk = 2048, D = 1024, H = 16;
  const float* query = (const float*)d_in[0];   // [B,Tq,D]
  const float* keyv  = (const float*)d_in[1];   // [B,Tk,D]
  const int*   maskp = (const int*)d_in[2];     // [B,1,Tq,Tk]
  const float* bias  = (const float*)d_in[3];   // [H,Tq,Tk]
  const float* Wq    = (const float*)d_in[4];   // [D,D]
  const float* Wkv   = (const float*)d_in[5];   // [D,2D]
  const float* Wo    = (const float*)d_in[6];   // [D,D]
  float* out = (float*)d_out;                   // [B,Tq,D]

  char* ws = (char*)d_ws;
  __bf16* q_bf    = (__bf16*)(ws);                           //  8 MB
  __bf16* kv_bf   = (__bf16*)(ws + (size_t)8  * 1024*1024);  // 16 MB
  __bf16* attn_bf = (__bf16*)(ws + (size_t)24 * 1024*1024);  //  8 MB

  const int M = Bb * Tq;  // 4096

  // 1) q = query @ Wq            (M x 1024, bf16)
  gemm_f32f32_bf16<<<dim3(D / 64, M / 128), 256, 0, stream>>>(
      query, Wq, q_bf, M, D, D);

  // 2) kv = key_value @ Wkv      (M x 2048, bf16): cols [0,1024)=K, [1024,2048)=V
  gemm_f32f32_bf16<<<dim3((2 * D) / 64, M / 128), 256, 0, stream>>>(
      keyv, Wkv, kv_bf, M, 2 * D, D);

  // 3) flash attention with fused bias + mask
  attention_flash<<<dim3(H, Tq / 16, Bb), 32, 0, stream>>>(
      q_bf, kv_bf, bias, maskp, attn_bf, Tq, Tk);

  // 4) out = attn @ Wo           (M x 1024, f32)
  gemm_bf16f32_f32<<<dim3(D / 64, M / 128), 256, 0, stream>>>(
      attn_bf, Wo, out, M, D, D);
}
